// GCNEncoder_2448131358806
// MI455X (gfx1250) — matrix operand order repaired and checked by reading, not compiled, not run on hardware
//
#include <hip/hip_runtime.h>
#include <hip/hip_bf16.h>

typedef __bf16 bf16_t;
typedef __attribute__((ext_vector_type(16))) __bf16 v16bf;
typedef __attribute__((ext_vector_type(8)))  float  v8f;

// ---------------------------------------------------------------------------
// Degree / normalization kernels
// ---------------------------------------------------------------------------
__global__ void deg_init_kernel(float* __restrict__ deg, int n) {
    int i = blockIdx.x * blockDim.x + threadIdx.x;
    if (i < n) deg[i] = 1.0f;  // self-loop contributes 1 to every node's degree
}

__global__ void deg_acc_kernel(const long long* __restrict__ dst,
                               float* __restrict__ deg, int e) {
    int i = blockIdx.x * blockDim.x + threadIdx.x;
    if (i < e) atomicAdd(&deg[(int)dst[i]], 1.0f);
}

__global__ void deg_rsqrt_kernel(float* __restrict__ deg, int n) {
    int i = blockIdx.x * blockDim.x + threadIdx.x;
    if (i < n) deg[i] = rsqrtf(deg[i]);  // deg >= 1 always (self-loop)
}

// ---------------------------------------------------------------------------
// GEMM: C[M,NOUT] = A[M,256] x W[256,NOUT] with v_wmma_f32_16x16x32_bf16,
// split-precision bf16x3: v = hi + lo, A*B ~= Alo*Bhi + Ahi*Blo + Ahi*Bhi
// (smallest terms accumulated first), f32 accumulator -> near-fp32 accuracy.
//
// One wave32 computes MT=5 vertical 16x16 tiles sharing one N-tile, so the
// strided B fragment (16 scalar loads + hi/lo split) is loaded once per
// k-step and reused across 5 A fragments / 15 WMMAs.
//
// Fragment layouts (wave32, 16-bit operands):
//   A: lane m = lane&15; VGPR v holds contiguous K pair at
//      kbase = (v&3)*2 + (lane>=16 ? 8 : 0) + (v>=4 ? 16 : 0)
//   B: lane n = lane&15; element j is K = (j&7) + (lane>=16 ? 8 : 0) + (j>=8 ? 16 : 0)
//   C/D: lane n = lane&15; VGPR r is row m = r + (lane>=16 ? 8 : 0)
// ---------------------------------------------------------------------------
template <int NOUT, int MT>
__global__ void gemm_wmma_bf16x3_kernel(const float* __restrict__ A,
                                        const float* __restrict__ W,
                                        float* __restrict__ C, int M) {
    constexpr int K = 256;
    constexpr int NTN = NOUT / 16;
    const int lane = threadIdx.x & 31;
    const int wave = threadIdx.x >> 5;
    const int tilesM = M >> 4;
    const int groupsM = tilesM / MT;          // M multiple of 16*MT by design

    const int grp = blockIdx.x * (blockDim.x >> 5) + wave;
    const int gm = grp / NTN;
    const int tn = grp % NTN;
    if (gm >= groupsM) return;  // wave-uniform: EXEC stays all-ones for WMMA

    const int hi   = lane >> 4;       // 0 for lanes 0-15, 1 for lanes 16-31
    const int l15  = lane & 15;
    const int ncol = tn * 16 + l15;
    const int m0   = gm * MT * 16;

    v8f acc[MT];
#pragma unroll
    for (int i = 0; i < MT; ++i) acc[i] = (v8f){};

    for (int k0 = 0; k0 < K; k0 += 32) {
        // ---- B fragment (shared by all MT tiles), hi/lo split ----
        v16bf bh, bl;
#pragma unroll
        for (int j = 0; j < 16; ++j) {
            const int kk = k0 + (j & 7) + (hi ? 8 : 0) + ((j >= 8) ? 16 : 0);
            const float w = W[(size_t)kk * NOUT + ncol];
            const bf16_t wh = (bf16_t)w;
            bh[j] = wh;
            bl[j] = (bf16_t)(w - (float)wh);
        }
#pragma unroll
        for (int i = 0; i < MT; ++i) {
            const float* __restrict__ arow = A + (size_t)(m0 + i * 16 + l15) * K;
            v16bf ah, al;
#pragma unroll
            for (int v = 0; v < 8; ++v) {
                const int kk = k0 + ((v & 3) * 2) + (hi ? 8 : 0) + ((v >= 4) ? 16 : 0);
                const float2 f2 = *reinterpret_cast<const float2*>(arow + kk);
                const bf16_t hx = (bf16_t)f2.x;
                const bf16_t hy = (bf16_t)f2.y;
                ah[2 * v]     = hx;
                ah[2 * v + 1] = hy;
                al[2 * v]     = (bf16_t)(f2.x - (float)hx);
                al[2 * v + 1] = (bf16_t)(f2.y - (float)hy);
            }
            // smallest partial products first, hi*hi last
            acc[i] = __builtin_amdgcn_wmma_f32_16x16x32_bf16(
                false, al, false, bh, (short)0, acc[i], false, false);
            acc[i] = __builtin_amdgcn_wmma_f32_16x16x32_bf16(
                false, ah, false, bl, (short)0, acc[i], false, false);
            acc[i] = __builtin_amdgcn_wmma_f32_16x16x32_bf16(
                false, ah, false, bh, (short)0, acc[i], false, false);
        }
    }

#pragma unroll
    for (int i = 0; i < MT; ++i) {
#pragma unroll
        for (int r = 0; r < 8; ++r) {
            const int mrow = m0 + i * 16 + r + (hi ? 8 : 0);
            C[(size_t)mrow * NOUT + ncol] = acc[i][r];
        }
    }
}

// ---------------------------------------------------------------------------
// out[i,:] = h[i,:] * dinv[i]^2   (self-loop term, initializes the accumulator)
// ---------------------------------------------------------------------------
template <int C>
__global__ void self_init_kernel(const float* __restrict__ h,
                                 const float* __restrict__ dinv,
                                 float* __restrict__ out, int n) {
    const int idx = blockIdx.x * blockDim.x + threadIdx.x;  // over n * C/4
    if (idx >= n * (C / 4)) return;
    const int i = idx / (C / 4);
    const float di = dinv[i];
    const float s = di * di;
    float4 v = reinterpret_cast<const float4*>(h)[idx];
    v.x *= s; v.y *= s; v.z *= s; v.w *= s;
    reinterpret_cast<float4*>(out)[idx] = v;
}

// ---------------------------------------------------------------------------
// Edge scatter: one wave32 per edge. c = lane + 32*j -> fully-coalesced 128B
// gathers of h[src] (L2-resident working set) and coalesced f32 atomic adds
// into out[dst] (L2 atomics).
// ---------------------------------------------------------------------------
template <int C>
__global__ void edge_scatter_kernel(const long long* __restrict__ src,
                                    const long long* __restrict__ dst,
                                    const float* __restrict__ dinv,
                                    const float* __restrict__ h,
                                    float* __restrict__ out, int e) {
    const int lane = threadIdx.x & 31;
    const int edge = (blockIdx.x * blockDim.x + threadIdx.x) >> 5;
    if (edge >= e) return;
    const int s = (int)src[edge];
    const int d = (int)dst[edge];
    const float norm = dinv[s] * dinv[d];
    const float* __restrict__ hs = h + (size_t)s * C;
    float* __restrict__ od = out + (size_t)d * C;
#pragma unroll
    for (int j = 0; j < C / 32; ++j) {
        const int c = lane + j * 32;
        atomicAdd(&od[c], hs[c] * norm);
    }
}

// ---------------------------------------------------------------------------
// out[i,c] = (in[i,c] + b[c])  with optional ReLU.  C is a power of two.
// ---------------------------------------------------------------------------
template <int C, bool RELU>
__global__ void bias_act_kernel(const float* __restrict__ in,
                                const float* __restrict__ b,
                                float* __restrict__ out, int total) {
    const int idx = blockIdx.x * blockDim.x + threadIdx.x;
    if (idx >= total) return;
    const int c = idx & (C - 1);
    float v = in[idx] + b[c];
    if (RELU) v = fmaxf(v, 0.0f);
    out[idx] = v;
}

// ---------------------------------------------------------------------------
// Launch
// ---------------------------------------------------------------------------
extern "C" void kernel_launch(void* const* d_in, const int* in_sizes, int n_in,
                              void* d_out, int out_size, void* d_ws, size_t ws_size,
                              hipStream_t stream) {
    constexpr int IN_C = 256, HID_C = 256, OUT_C = 128;
    constexpr int MT = 5;  // 50000/16 = 3125 = 5^5 -> groups of 5 tiles, no tail

    const float*     x  = (const float*)d_in[0];
    const long long* ei = (const long long*)d_in[1];  // int64 [2, E]
    const float*     W1 = (const float*)d_in[2];
    const float*     b1 = (const float*)d_in[3];
    const float*     W2 = (const float*)d_in[4];
    const float*     b2 = (const float*)d_in[5];
    float*           out = (float*)d_out;

    const int N = in_sizes[0] / IN_C;   // 50000
    const int E = in_sizes[1] / 2;      // 800000
    const long long* src = ei;
    const long long* dst = ei + E;

    // Workspace layout: dinv | bufH (N*256 f32) | bufA (N*256 f32)
    char* ws = (char*)d_ws;
    const size_t off1 = ((size_t)N * sizeof(float) + 255) & ~(size_t)255;
    float* dinv = (float*)ws;
    float* bufH = (float*)(ws + off1);
    float* bufA = (float*)(ws + off1 + (size_t)N * HID_C * sizeof(float));

    const int T = 256;
    // --- normalization ---
    deg_init_kernel<<<(N + T - 1) / T, T, 0, stream>>>(dinv, N);
    deg_acc_kernel<<<(E + T - 1) / T, T, 0, stream>>>(dst, dinv, E);
    deg_rsqrt_kernel<<<(N + T - 1) / T, T, 0, stream>>>(dinv, N);

    // --- layer 1: h = x @ W1 ---
    {
        const int groups = (N / (16 * MT)) * (HID_C / 16);   // 625 * 16
        gemm_wmma_bf16x3_kernel<HID_C, MT><<<(groups + 3) / 4, 128, 0, stream>>>(x, W1, bufH, N);
    }
    self_init_kernel<HID_C><<<(N * (HID_C / 4) + T - 1) / T, T, 0, stream>>>(bufH, dinv, bufA, N);
    edge_scatter_kernel<HID_C><<<((E * 32) + T - 1) / T, T, 0, stream>>>(src, dst, dinv, bufH, bufA, E);
    bias_act_kernel<HID_C, true><<<((N * HID_C) + T - 1) / T, T, 0, stream>>>(bufA, b1, bufH, N * HID_C);

    // --- layer 2: h2 = relu_out @ W2 ---
    {
        const int groups = (N / (16 * MT)) * (OUT_C / 16);   // 625 * 8
        gemm_wmma_bf16x3_kernel<OUT_C, MT><<<(groups + 3) / 4, 128, 0, stream>>>(bufH, W2, bufA, N);
    }
    self_init_kernel<OUT_C><<<(N * (OUT_C / 4) + T - 1) / T, T, 0, stream>>>(bufA, dinv, out, N);
    edge_scatter_kernel<OUT_C><<<((E * 32) + T - 1) / T, T, 0, stream>>>(src, dst, dinv, bufA, out, E);
    bias_act_kernel<OUT_C, false><<<((N * OUT_C) + T - 1) / T, T, 0, stream>>>(out, b2, out, N * OUT_C);
}
